// ENGRU_18588618457510
// MI455X (gfx1250) — matrix operand order repaired and checked by reading, not compiled, not run on hardware
//
#include <hip/hip_runtime.h>

typedef __attribute__((ext_vector_type(16))) __bf16 v16bf;
typedef __attribute__((ext_vector_type(8)))  float  v8f;
typedef __attribute__((ext_vector_type(4)))  unsigned v4u;
typedef __attribute__((ext_vector_type(8)))  int      v8i_;
typedef __attribute__((ext_vector_type(4)))  int      v4i_;

#define HH 256     // hidden size (compile-time specialized)
#define GG 768     // 3*H
#define BSLICE 16  // batches per recurrence workgroup (one WMMA M-tile)

#if __has_builtin(__builtin_amdgcn_tensor_load_to_lds) && \
    __has_builtin(__builtin_amdgcn_s_wait_tensorcnt)
#define HAVE_TDM 1
#else
#define HAVE_TDM 0
#endif

// ---------------- kernel A: fp32 -> bf16 conversion ----------------
__global__ void gru_cvt_bf16(const float* __restrict__ s, __bf16* __restrict__ d, int n) {
  int i = blockIdx.x * blockDim.x + threadIdx.x;
  if (i < n) d[i] = (__bf16)s[i];
}

#if HAVE_TDM
// Issue one 2D TDM tile load (16-bit elements): tile_d1 rows x tile_d0 elems,
// row stride stride0 (elements), into LDS at lds_off. Uniform per wave.
// D# packed per CDNA5 ISA ch.8 (group0: count/lds/global/type, group1: dims).
__device__ __forceinline__ void tdm_load_2d_b16(unsigned lds_off, const void* gaddr,
                                                unsigned tensor_d0, unsigned tensor_d1,
                                                unsigned tile_d0, unsigned tile_d1,
                                                unsigned stride0) {
  unsigned long long ga = (unsigned long long)(uintptr_t)gaddr;
  v4u g0;
  g0[0] = 1u;                                             // count=1, user mode
  g0[1] = lds_off;                                        // lds_addr (bytes)
  g0[2] = (unsigned)(ga & 0xffffffffu);                   // global_addr[31:0]
  g0[3] = (unsigned)((ga >> 32) & 0x01ffffffu) | (2u << 30); // addr[56:32] | type=2
  v8i_ g1;
  g1[0] = (int)(1u << 16);                                // data_size=1 (2 bytes)
  g1[1] = (int)((tensor_d0 & 0xffffu) << 16);             // tensor_dim0[15:0]
  g1[2] = (int)(((tensor_d0 >> 16) & 0xffffu) |
                ((tensor_d1 & 0xffffu) << 16));           // dim0[31:16] | dim1[15:0]
  g1[3] = (int)(((tensor_d1 >> 16) & 0xffffu) |
                ((tile_d0 & 0xffffu) << 16));             // dim1[31:16] | tile_dim0
  g1[4] = (int)(tile_d1 & 0xffffu);                       // tile_dim1 (tile_dim2=0)
  g1[5] = (int)stride0;                                   // tensor_dim0_stride[31:0]
  g1[6] = 0;
  g1[7] = 0;
  v4i_ z4 = {0, 0, 0, 0};
#if defined(__clang_major__) && (__clang_major__ >= 23)
  v8i_ z8 = {0, 0, 0, 0, 0, 0, 0, 0};
  __builtin_amdgcn_tensor_load_to_lds(g0, g1, z4, z4, z8, 0);
#else
  __builtin_amdgcn_tensor_load_to_lds(g0, g1, z4, z4, 0);
#endif
}
#endif

// ---------------- kernel B: xg[t][b][g] = x_bf[b*T+t,:] . W_ih[g,:] + b_ih[g] ----------
// WMMA GEMM, M = B*T, N = G, K = D. Block tile 64x64, 4 waves of 32x32.
// Tiles are pure bf16 byte copies -> staged by the Tensor Data Mover.
__global__ __launch_bounds__(128)
void gru_xproj(const __bf16* __restrict__ xbf, const __bf16* __restrict__ wih,
               const float* __restrict__ bih, float* __restrict__ xg,
               int Bsz, int T, int D, int G) {
  __shared__ __bf16 As[64][32];  // [m][k]
  __shared__ __bf16 Bs[64][32];  // [n][k]

  const int m0    = blockIdx.x * 64;
  const int n0blk = blockIdx.y * 64;
  const int tid   = threadIdx.x;   // 0..127
  const int wave  = tid >> 5;
  const int lane  = tid & 31;
  const int wm    = (wave & 1) * 32;
  const int wn    = (wave >> 1) * 32;
  const int lhalf = lane >> 4;
  const int lmod  = lane & 15;

  v8f c[2][2] = {};

  for (int k0 = 0; k0 < D; k0 += 32) {
#if HAVE_TDM
    if (tid < 32) {  // wave 0 drives the TDM; EXEC-independent scalar-descriptor op
      tdm_load_2d_b16((unsigned)(uintptr_t)&As[0][0],
                      xbf + (size_t)m0 * D + k0,
                      (unsigned)D, (unsigned)(Bsz * T), 32u, 64u, (unsigned)D);
      tdm_load_2d_b16((unsigned)(uintptr_t)&Bs[0][0],
                      wih + (size_t)n0blk * D + k0,
                      (unsigned)D, (unsigned)G, 32u, 64u, (unsigned)D);
      __builtin_amdgcn_s_wait_tensorcnt(0);
    }
    __syncthreads();
#else
    {  // fallback: cooperative 32B-per-thread vector copies
      int row = tid >> 1;
      int kk  = (tid & 1) * 16;
      *(v16bf*)&As[row][kk] = *(const v16bf*)(xbf + (size_t)(m0 + row) * D + k0 + kk);
      *(v16bf*)&Bs[row][kk] = *(const v16bf*)(wih + (size_t)(n0blk + row) * D + k0 + kk);
    }
    __syncthreads();
#endif

    // Fragments per ISA layout; 2x2 WMMA tiles per wave
    v16bf a[2], b[2];
    #pragma unroll
    for (int im = 0; im < 2; ++im) {
      int m = wm + im * 16 + lmod;
      #pragma unroll
      for (int i = 0; i < 8; ++i) {
        a[im][i]     = As[m][8 * lhalf + i];        // K = 8h + i
        a[im][8 + i] = As[m][16 + 8 * lhalf + i];   // K = 16 + 8h + i
      }
    }
    #pragma unroll
    for (int in = 0; in < 2; ++in) {
      int n = wn + in * 16 + lmod;
      #pragma unroll
      for (int i = 0; i < 16; ++i) b[in][i] = Bs[n][8 * lhalf + (i & 7) + 16 * (i >> 3)];  // K = 8h + i (i<8), 16 + 8h + (i-8) (i>=8)
    }
    #pragma unroll
    for (int im = 0; im < 2; ++im)
      #pragma unroll
      for (int in = 0; in < 2; ++in)
        c[im][in] = __builtin_amdgcn_wmma_f32_16x16x32_bf16(
            false, a[im], false, b[in], (short)0, c[im][in], false, false);
    __syncthreads();
  }

  // Epilogue: one division per block (T % 64 == 0 -> whole block shares b)
  const int bi  = m0 / T;
  const int tt0 = m0 - bi * T;
  #pragma unroll
  for (int im = 0; im < 2; ++im) {
    #pragma unroll
    for (int in = 0; in < 2; ++in) {
      int gcol   = n0blk + wn + in * 16 + lmod;
      float bias = bih[gcol];
      #pragma unroll
      for (int v = 0; v < 8; ++v) {
        int mloc = wm + im * 16 + v + 8 * lhalf;
        xg[((size_t)(tt0 + mloc) * Bsz + bi) * G + gcol] = c[im][in][v] + bias;
      }
    }
  }
}

// ---------------- kernel C: sequential GRU recurrence ----------------
// Batch-partitioned persistent workgroups (recurrence independent per batch row).
// 512 threads = 16 waves; wave w owns gate tiles {w, 16+w, 32+w} so the r/z/n
// columns [16w,16w+16) stay in-wave and the pointwise update reads accumulators.
__global__ __launch_bounds__(512)
void gru_recur(const float* __restrict__ xg,   // [T][B][G] fp32
               const float* __restrict__ h0,   // [B][H]
               const __bf16* __restrict__ whh, // [G][H] bf16 (L2-resident)
               const float* __restrict__ bhh,  // [G]
               float* __restrict__ out,        // hs [B][T][H] ++ hT [B][H]
               int Bsz, int T) {
  __shared__ float  hf[BSLICE][HH];   // fp32 carried hidden state
  __shared__ __bf16 hb[BSLICE][HH];   // bf16 copy read as WMMA A fragments

  const int b0  = blockIdx.x * BSLICE;
  const int tid = threadIdx.x;

  for (int e = tid; e < BSLICE * HH; e += blockDim.x) {
    int m = e >> 8, j = e & (HH - 1);
    float v = h0[(size_t)(b0 + m) * HH + j];
    hf[m][j] = v;
    hb[m][j] = (__bf16)v;
  }
  __syncthreads();

  const int wave  = tid >> 5;          // 0..15
  const int lane  = tid & 31;
  const int lhalf = lane >> 4;
  const int lmod  = lane & 15;
  const int jcol  = wave * 16 + lmod;  // hidden column this lane updates

  const float bh_r = bhh[jcol];
  const float bh_z = bhh[HH + jcol];
  const float bh_n = bhh[2 * HH + jcol];

  for (int t = 0; t < T; ++t) {
    const float* xgt = xg + ((size_t)t * Bsz + b0) * GG;

    if (t + 1 < T)  // cover next step's 48KB xg slice: 512 thr * 96B
      __builtin_prefetch(xg + ((size_t)(t + 1) * Bsz + b0) * GG + (size_t)tid * 24, 0, 0);

    // hg(16 x 768) = h(16 x 256) . W_hh^T ; this wave's 3 gate tiles
    v8f c[3] = {};
    #pragma unroll
    for (int kk = 0; kk < HH / 32; ++kk) {
      v16bf a;
      int kb = kk * 32 + 16 * lhalf;
      #pragma unroll
      for (int i = 0; i < 8; ++i) {
        a[i]     = hb[lmod][kb + i];
        a[8 + i] = hb[lmod][kb + 8 + i];
      }
      #pragma unroll
      for (int nt = 0; nt < 3; ++nt) {
        int n = (nt * 16 + wave) * 16 + lmod;   // gate tiles w, 16+w, 32+w
        v16bf b = *(const v16bf*)(whh + (size_t)n * HH + kk * 32 + 16 * lhalf);
        c[nt] = __builtin_amdgcn_wmma_f32_16x16x32_bf16(
            false, a, false, b, (short)0, c[nt], false, false);
      }
    }
    __syncthreads();  // all waves done reading hb before it is overwritten

    // In-wave pointwise GRU update: lane (lmod,lhalf), vgpr v -> (m, jcol)
    #pragma unroll
    for (int v = 0; v < 8; ++v) {
      int m = v + 8 * lhalf;
      float xr = xgt[m * GG + jcol];
      float xz = xgt[m * GG + HH + jcol];
      float xn = xgt[m * GG + 2 * HH + jcol];
      float r = 1.0f / (1.0f + __expf(-(xr + c[0][v] + bh_r)));
      float z = 1.0f / (1.0f + __expf(-(xz + c[1][v] + bh_z)));
      float n = tanhf(xn + r * (c[2][v] + bh_n));
      float hnew = (1.0f - z) * n + z * hf[m][jcol];
      hf[m][jcol] = hnew;
      hb[m][jcol] = (__bf16)hnew;
      out[((size_t)(b0 + m) * T + t) * HH + jcol] = hnew;
    }
    __syncthreads();  // hb updated before next step's A-fragment reads
  }

  float* hT = out + (size_t)Bsz * T * HH;
  for (int e = tid; e < BSLICE * HH; e += blockDim.x) {
    int m = e >> 8, j = e & (HH - 1);
    hT[(size_t)(b0 + m) * HH + j] = hf[m][j];
  }
}

// ---------------- launch ----------------
extern "C" void kernel_launch(void* const* d_in, const int* in_sizes, int n_in,
                              void* d_out, int out_size, void* d_ws, size_t ws_size,
                              hipStream_t stream) {
  (void)n_in; (void)out_size; (void)ws_size;
  const float* x    = (const float*)d_in[0];
  const float* h0   = (const float*)d_in[1];
  const float* W_ih = (const float*)d_in[2];
  const float* W_hh = (const float*)d_in[3];
  const float* b_ih = (const float*)d_in[4];
  const float* b_hh = (const float*)d_in[5];

  const int H   = in_sizes[5] / 3;            // 256
  const int G   = 3 * H;                      // 768
  const int D   = in_sizes[2] / G;            // 256
  const int Bsz = in_sizes[1] / H;            // 64
  const int T   = in_sizes[0] / (Bsz * D);    // 2048

  // workspace layout: [wih bf16][whh bf16][x bf16][xg fp32]
  char* ws = (char*)d_ws;
  size_t o1 = ((size_t)G * D * sizeof(__bf16) + 255) & ~(size_t)255;
  size_t o2 = (o1 + (size_t)G * H * sizeof(__bf16) + 255) & ~(size_t)255;
  size_t o3 = (o2 + (size_t)Bsz * T * D * sizeof(__bf16) + 255) & ~(size_t)255;
  __bf16* wih_bf = (__bf16*)ws;
  __bf16* whh_bf = (__bf16*)(ws + o1);
  __bf16* xbf    = (__bf16*)(ws + o2);
  float*  xg     = (float*)(ws + o3);
  float*  out    = (float*)d_out;

  {
    int n = G * D;
    gru_cvt_bf16<<<(n + 255) / 256, 256, 0, stream>>>(W_ih, wih_bf, n);
  }
  {
    int n = G * H;
    gru_cvt_bf16<<<(n + 255) / 256, 256, 0, stream>>>(W_hh, whh_bf, n);
  }
  {
    int n = Bsz * T * D;
    gru_cvt_bf16<<<(n + 255) / 256, 256, 0, stream>>>(x, xbf, n);
  }
  {
    dim3 grid((Bsz * T) / 64, G / 64);
    gru_xproj<<<grid, 128, 0, stream>>>(xbf, wih_bf, b_ih, xg, Bsz, T, D, G);
  }
  {
    gru_recur<<<Bsz / BSLICE, 512, 0, stream>>>(xg, h0, whh_bf, b_hh, out, Bsz, T);
  }
}
